// Linear2Bit_31559419691550
// MI455X (gfx1250) — matrix operand order repaired and checked by reading, not compiled, not run on hardware
//
#include <hip/hip_runtime.h>
#include <stdint.h>

// ---------------------------------------------------------------------------
// 2-bit quantized linear: y = (x @ Wq^T) * scale + bias
// fp32->f16 / int8->f16 (exact) prep into workspace, then WMMA f16 GEMM with
// f32 accumulation. Global->LDS staging via CDNA5 async-to-LDS copies
// (ASYNCcnt), triple-buffered dynamic-LDS ring, 64x64 per-wave tiles.
// All 16 fragment ds_load_b128 issue at stage top so LDS latency hides
// behind the 16 v_wmma instructions.
// ---------------------------------------------------------------------------

typedef __attribute__((ext_vector_type(16))) _Float16 v16h;
typedef __attribute__((ext_vector_type(8)))  _Float16 v8h;
typedef __attribute__((ext_vector_type(8)))  float    v8f;
typedef __attribute__((ext_vector_type(4)))  float    v4f;

#define TOKENS 8192
#define INF    4096
#define OUTF   4096

#define BM 256
#define BN 128
#define BK 32
#define LDW 40                        // padded LDS row (f16): 80B stride
#define A_F16 (BM * LDW)              // 10240 f16
#define B_F16 (BN * LDW)              // 5120 f16
#define STAGE_F16 (A_F16 + B_F16)     // 15360 f16 = 30720 B / stage
#define NSTAGE 3
#define LDS_BYTES (NSTAGE * STAGE_F16 * 2)   // 92160 B (dynamic LDS)
#define KSTEPS (INF / BK)             // 128

// ---------------- prep: fp32 -> f16 -------------------------------------
__global__ __launch_bounds__(256) void l2b_cvt_x(const float* __restrict__ x,
                                                 _Float16* __restrict__ xh,
                                                 int n) {
    int idx = (blockIdx.x * 256 + threadIdx.x) * 8;
    if (idx >= n) return;
    const v4f* p = (const v4f*)(x + idx);
    v4f f0 = p[0];
    v4f f1 = p[1];
    v8h o;
    o[0] = (_Float16)f0[0]; o[1] = (_Float16)f0[1];
    o[2] = (_Float16)f0[2]; o[3] = (_Float16)f0[3];
    o[4] = (_Float16)f1[0]; o[5] = (_Float16)f1[1];
    o[6] = (_Float16)f1[2]; o[7] = (_Float16)f1[3];
    *(v8h*)(xh + idx) = o;
}

// ---------------- prep: int8 {-2..1} -> f16 (exact) ----------------------
__global__ __launch_bounds__(256) void l2b_cvt_w(const int8_t* __restrict__ w,
                                                 _Float16* __restrict__ wh,
                                                 int n) {
    int idx = (blockIdx.x * 256 + threadIdx.x) * 8;
    if (idx >= n) return;
    uint64_t raw = *(const uint64_t*)(w + idx);
    v8h o;
#pragma unroll
    for (int i = 0; i < 8; ++i) {
        int8_t b = (int8_t)(raw >> (8 * i));
        o[i] = (_Float16)(float)b;
    }
    *(v8h*)(wh + idx) = o;
}

// ---------------- main WMMA GEMM -----------------------------------------
__global__ __launch_bounds__(256) void l2b_gemm(const _Float16* __restrict__ xh,
                                                const _Float16* __restrict__ wh,
                                                const float* __restrict__ scale_p,
                                                const float* __restrict__ bias,
                                                float* __restrict__ out) {
    extern __shared__ _Float16 lds[];   // NSTAGE * 30720 B ring

    const int tid  = threadIdx.x;
    const int lane = tid & 31;
    const int wave = tid >> 5;          // 0..7
    const int wm   = wave >> 1;         // 0..3 -> 64-row slab
    const int wn   = wave & 1;          // 0..1 -> 64-col slab

    const int m0 = blockIdx.y * BM;
    const int n0 = blockIdx.x * BN;

    const float scl = scale_p[0];

    // SGPR panel bases for GVS-mode async copies
    const uint64_t aBase = (uint64_t)(uintptr_t)(xh + (size_t)m0 * INF);
    const uint64_t bBase = (uint64_t)(uintptr_t)(wh + (size_t)n0 * INF);

    // Async copy mapping. Per stage/thread: A = 4x16B, B = 2x16B.
    // chunk = tid + 256*i ; row = chunk>>2 ; cc = chunk&3 (8-f16 chunk)
    uint32_t voffA[4], voffB[2];
    uint32_t ldsOffA[4], ldsOffB[2];
#pragma unroll
    for (int i = 0; i < 4; ++i) {
        int chunk = tid + 256 * i;
        int row = chunk >> 2;           // 0..255
        int cc  = chunk & 3;
        voffA[i]   = (uint32_t)(row * (INF * 2) + cc * 16);
        ldsOffA[i] = (uint32_t)(row * (LDW * 2) + cc * 16);
    }
#pragma unroll
    for (int i = 0; i < 2; ++i) {
        int chunk = tid + 256 * i;
        int row = chunk >> 2;           // 0..127
        int cc  = chunk & 3;
        voffB[i]   = (uint32_t)(row * (INF * 2) + cc * 16);
        ldsOffB[i] = (uint32_t)(A_F16 * 2 + row * (LDW * 2) + cc * 16);
    }
    const uint32_t ldsBase = (uint32_t)(uintptr_t)(&lds[0]);

    // Issue one stage (6 async instructions / thread), then advance offsets.
    auto issue = [&](int stage) {
        uint32_t sb = ldsBase + (uint32_t)stage * (STAGE_F16 * 2);
#pragma unroll
        for (int i = 0; i < 4; ++i) {
            asm volatile("global_load_async_to_lds_b128 %0, %1, %2"
                         :: "v"(sb + ldsOffA[i]), "v"(voffA[i]), "s"(aBase)
                         : "memory");
            voffA[i] += BK * 2;
        }
#pragma unroll
        for (int i = 0; i < 2; ++i) {
            asm volatile("global_load_async_to_lds_b128 %0, %1, %2"
                         :: "v"(sb + ldsOffB[i]), "v"(voffB[i]), "s"(bBase)
                         : "memory");
            voffB[i] += BK * 2;
        }
    };

    const v8f vzero = {};
    v8f acc[4][4];
#pragma unroll
    for (int i = 0; i < 4; ++i)
#pragma unroll
        for (int j = 0; j < 4; ++j) acc[i][j] = vzero;

    issue(0);
    issue(1);
    int sCur = 0;
    int sIss = 2;

    const int lm = lane & 15;
    const int hi = lane >> 4;           // 0 or 1 (half-wave)

    for (int kt = 0; kt < KSTEPS; ++kt) {
        // Oldest stage's 6 copies retired  <=>  per-wave ASYNCcnt <= 6
        asm volatile("s_wait_asynccnt 6" ::: "memory");
        __syncthreads();                // cross-wave visibility of stage kt

        if (kt + 2 < KSTEPS) issue(sIss);   // prefetch depth 2

        const _Float16* la = lds + sCur * STAGE_F16;
        const _Float16* lb = la + A_F16;

        // All fragment loads up-front: 8 B-loads + 8 A-loads in flight
        // before the first WMMA, so dscnt waits are satisfied in-stream.
        v16h bfr[4];
#pragma unroll
        for (int jn = 0; jn < 4; ++jn) {
            int n = wn * 64 + jn * 16 + lm;
            const _Float16* base = lb + n * LDW + hi * 8;
            union { v16h v; v8h h[2]; } u;
            u.h[0] = *(const v8h*)(base);
            u.h[1] = *(const v8h*)(base + 16);
            bfr[jn] = u.v;
        }
        v16h afr[4];
#pragma unroll
        for (int im = 0; im < 4; ++im) {
            const _Float16* base = la + (wm * 64 + im * 16 + lm) * LDW + hi * 8;
            union { v16h v; v8h h[2]; } u;
            u.h[0] = *(const v8h*)(base);
            u.h[1] = *(const v8h*)(base + 16);
            afr[im] = u.v;
        }
#pragma unroll
        for (int im = 0; im < 4; ++im)
#pragma unroll
            for (int jn = 0; jn < 4; ++jn)
                acc[im][jn] = __builtin_amdgcn_wmma_f32_16x16x32_f16(
                    false, afr[im], false, bfr[jn],
                    (short)0, acc[im][jn], false, false);

        if (++sCur == NSTAGE) sCur = 0;
        if (++sIss == NSTAGE) sIss = 0;
    }

    // epilogue: out = acc * scale + bias
#pragma unroll
    for (int im = 0; im < 4; ++im) {
#pragma unroll
        for (int jn = 0; jn < 4; ++jn) {
            int col = n0 + wn * 64 + jn * 16 + lm;
            float bv = bias[col];
#pragma unroll
            for (int v = 0; v < 8; ++v) {
                int row = m0 + wm * 64 + im * 16 + v + hi * 8;
                out[(size_t)row * OUTF + col] = acc[im][jn][v] * scl + bv;
            }
        }
    }
}

// ---------------------------------------------------------------------------
extern "C" void kernel_launch(void* const* d_in, const int* in_sizes, int n_in,
                              void* d_out, int out_size, void* d_ws, size_t ws_size,
                              hipStream_t stream) {
    (void)in_sizes; (void)n_in; (void)out_size; (void)ws_size;

    const float*   x     = (const float*)d_in[0];
    const int8_t*  wq    = (const int8_t*)d_in[1];
    const float*   scale = (const float*)d_in[2];
    const float*   bias  = (const float*)d_in[3];
    float*         out   = (float*)d_out;

    _Float16* xh = (_Float16*)d_ws;                       // 64 MB
    _Float16* wh = xh + (size_t)TOKENS * INF;             // +32 MB

    const int nx = TOKENS * INF;
    const int nw = OUTF * INF;
    l2b_cvt_x<<<nx / (256 * 8), 256, 0, stream>>>(x, xh, nx);
    l2b_cvt_w<<<nw / (256 * 8), 256, 0, stream>>>(wq, wh, nw);

    dim3 grid(OUTF / BN, TOKENS / BM);   // (32, 32)
    l2b_gemm<<<grid, 256, LDS_BYTES, stream>>>(xh, wh, scale, bias, out);
}
